// BiLSTM_CRF_9251359556277
// MI455X (gfx1250) — compile-verified, matrix operand order checked by reading, not run on hardware
//
#include <hip/hip_runtime.h>
#include <hip/hip_bf16.h>
#include <math.h>

// ---------------------------------------------------------------------------
// BiLSTM-CRF forward for MI455X (gfx1250, wave32).
// Per-timestep fused GEMM on v_wmma_f32_16x16x32_bf16:
//   gates[128,1024] = [x_t | h_{t-1}] (128x512 bf16) @ Wcat^T (512x1024 bf16)
// Weights live in LDS (fragment-packed, 256KB of the WGP's 320KB); cell state
// c and all gate activations stay in VGPRs for all 512 steps.
// ---------------------------------------------------------------------------

#define Bdim  128
#define Sdim  512
#define Edim  256
#define HDdim 256
#define Tdim  9
#define KCAT  512          // E + HD
#define NGATE 1024         // 4*HD
#define NBLK  8            // 4 blocks per direction
#define NBDIR 4            // blocks per direction (barrier fan-in)

typedef __attribute__((ext_vector_type(16))) __bf16 v16bf;
typedef __attribute__((ext_vector_type(8)))  __bf16 v8bf;
typedef __attribute__((ext_vector_type(8)))  float  v8f;

__device__ __forceinline__ float sigmoidf_(float x) {
    return 1.0f / (1.0f + __expf(-x));
}

// Load one 16x32 bf16 fragment (A operand) from a row-major matrix.
// ISA layout (CDNA5 ISA): lane L (r=L%16, hi=L/16) holds row r,
// K elements {hi*8..hi*8+7} and {16+hi*8..16+hi*8+7}  -> two 16B loads.
__device__ __forceinline__ v16bf load_frag(const __bf16* __restrict__ base,
                                           int row0, size_t ld, int k0,
                                           int r, int hi) {
    const __bf16* p = base + (size_t)(row0 + r) * ld + (size_t)(k0 + hi * 8);
    v8bf lo = *reinterpret_cast<const v8bf*>(p);
    v8bf hh = *reinterpret_cast<const v8bf*>(p + 16);
    return __builtin_shufflevector(lo, hh, 0,1,2,3,4,5,6,7,8,9,10,11,12,13,14,15);
}

// B fragment from fragment-packed LDS (plain loads -> ds_load_b128 after
// address-space inference; the step loop's fences keep them un-hoisted).
__device__ __forceinline__ v16bf lds_frag(const __bf16* WL, int fid, int lane) {
    const v8bf* p = reinterpret_cast<const v8bf*>(WL + (size_t)fid * 512 + lane * 16);
    v8bf lo = p[0];
    v8bf hh = p[1];
    return __builtin_shufflevector(lo, hh, 0,1,2,3,4,5,6,7,8,9,10,11,12,13,14,15);
}

#define WMMA_BF16(A, B, C) \
    __builtin_amdgcn_wmma_f32_16x16x32_bf16(false, (A), false, (B), (short)0, (C), false, false)

// ---------------------------------------------------------------------------
// Prep kernels (run every launch; also reset barrier/state for graph replay)
// ---------------------------------------------------------------------------
__global__ void prep_weights(const float* __restrict__ w_ih_f, const float* __restrict__ w_hh_f,
                             const float* __restrict__ b_ih_f, const float* __restrict__ b_hh_f,
                             const float* __restrict__ w_ih_b, const float* __restrict__ w_hh_b,
                             const float* __restrict__ b_ih_b, const float* __restrict__ b_hh_b,
                             __bf16* __restrict__ Wf, __bf16* __restrict__ Wb,
                             float* __restrict__ bias_f, float* __restrict__ bias_b) {
    int idx = blockIdx.x * blockDim.x + threadIdx.x;
    const int total = NGATE * KCAT;
    for (int i = idx; i < total; i += gridDim.x * blockDim.x) {
        int n = i >> 9;        // gate row 0..1023
        int k = i & 511;       // concat col 0..511
        float vf = (k < Edim) ? w_ih_f[n * Edim + k] : w_hh_f[n * HDdim + (k - Edim)];
        float vb = (k < Edim) ? w_ih_b[n * Edim + k] : w_hh_b[n * HDdim + (k - Edim)];
        Wf[i] = (__bf16)vf;
        Wb[i] = (__bf16)vb;
    }
    if (idx < NGATE) {
        bias_f[idx] = b_ih_f[idx] + b_hh_f[idx];
        bias_b[idx] = b_ih_b[idx] + b_hh_b[idx];
    }
}

__global__ void prep_embed(const int* __restrict__ X, const float* __restrict__ emb,
                           __bf16* __restrict__ x_bf) {
    size_t idx = (size_t)blockIdx.x * blockDim.x + threadIdx.x;
    const size_t total = (size_t)Bdim * Sdim * Edim / 4;
    if (idx >= total) return;
    size_t e4 = idx & (Edim / 4 - 1);
    size_t bs = idx / (Edim / 4);
    int tok = X[bs];
    const float4 v = *reinterpret_cast<const float4*>(emb + (size_t)tok * Edim + e4 * 4);
    __bf16* o = x_bf + bs * Edim + e4 * 4;
    o[0] = (__bf16)v.x; o[1] = (__bf16)v.y; o[2] = (__bf16)v.z; o[3] = (__bf16)v.w;
}

__global__ void prep_state(__bf16* __restrict__ Hbuf, unsigned* __restrict__ bar) {
    int idx = blockIdx.x * blockDim.x + threadIdx.x;
    const int total = 2 * 2 * Bdim * HDdim;   // [pingpong][dir][B][HD]
    if (idx < total) Hbuf[idx] = (__bf16)0.0f;
    if (idx < 64) bar[idx] = 0u;              // both direction counters (+ padding)
}

// ---------------------------------------------------------------------------
// Persistent recurrent kernel: 8 blocks (4 fwd + 4 bwd), 512 thr = 16 waves.
// Block d owns hidden cols [64d, 64d+64). Wave (wm = wave&3, wq = wave>>2):
//   M tiles {2wm, 2wm+1} (batch rows 32wm..32wm+31)
//   x all 4 gates for cols [64d+16wq, +16)
// => i/f/g/o for one (b,j) sit in one lane's accumulators; c stays in VGPRs.
// W slice (256KB) fragment-packed in LDS; B frags reused across 2 M tiles,
// A frags across 4 gates, with a one-deep A-preload rotation so the global
// load latency is covered by the previous k-step's 8 WMMAs + ds loads.
// Per-direction spin barrier (4 arrivals) on separate cachelines.
// ---------------------------------------------------------------------------
__global__ __launch_bounds__(512, 1)
void lstm_recurrent(const __bf16* __restrict__ x_bf,
                    const __bf16* __restrict__ Wf, const __bf16* __restrict__ Wb,
                    const float* __restrict__ bias_f, const float* __restrict__ bias_b,
                    __bf16* __restrict__ Hbuf,
                    __bf16* __restrict__ hs_f, __bf16* __restrict__ hs_b,
                    unsigned* __restrict__ bar) {
    extern __shared__ __bf16 WL[];     // 4 gates * 4 wq * 16 kk frags * 512 = 256KB

    const int tid  = threadIdx.x;
    const int wave = tid >> 5;
    const int lane = tid & 31;
    const int wm = wave & 3;           // covers M tiles {2wm, 2wm+1}
    const int wq = wave >> 2;          // 16-col quarter 0..3
    const int r  = lane & 15;
    const int hi = lane >> 4;
    const int dir = blockIdx.x >> 2;   // 0 = fwd, 1 = bwd
    const int d   = blockIdx.x & 3;
    const int j0  = d * 64;

    const __bf16* W    = dir ? Wb : Wf;
    const float*  bias = dir ? bias_b : bias_f;
    __bf16*       hs   = dir ? hs_b : hs_f;
    unsigned*     barD = bar + dir * 16;       // 64B apart

    // ---- stage this block's W slice into LDS, fragment-packed ----
    if (wm == 0) {                      // waves 0,4,8,12 stage their wq
        for (int kk = 0; kk < 16; ++kk) {
            for (int g = 0; g < 4; ++g) {
                const __bf16* p = W + (size_t)(g * HDdim + j0 + 16 * wq + r) * KCAT
                                    + (size_t)(kk * 32 + hi * 8);
                v8bf lo = *reinterpret_cast<const v8bf*>(p);
                v8bf hh = *reinterpret_cast<const v8bf*>(p + 16);
                __bf16* dst = WL + (size_t)(((g * 4 + wq) * 16 + kk)) * 512 + lane * 16;
                *reinterpret_cast<v8bf*>(dst)     = lo;
                *reinterpret_cast<v8bf*>(dst + 8) = hh;
            }
        }
    }
    __syncthreads();

    const int jcol = j0 + 16 * wq + r;     // this lane's hidden column
    float bg[4];
#pragma unroll
    for (int g = 0; g < 4; ++g) bg[g] = bias[g * HDdim + jcol];

    float c[2][8];
#pragma unroll
    for (int mt = 0; mt < 2; ++mt)
#pragma unroll
        for (int v = 0; v < 8; ++v) c[mt][v] = 0.0f;

    const int brow0 = 32 * wm;
    const int brow1 = 32 * wm + 16;
    const size_t SE = (size_t)Sdim * Edim;
    const v8f vzero = {0.f, 0.f, 0.f, 0.f, 0.f, 0.f, 0.f, 0.f};

    for (int st = 0; st < Sdim; ++st) {
        const int t = dir ? (Sdim - 1 - st) : st;
        const __bf16* Xbase = x_bf + (size_t)t * Edim;   // batch-row stride S*E
        const __bf16* Hprev = Hbuf + (size_t)(((st & 1) * 2) + dir) * (Bdim * HDdim);
        __bf16*       Hnext = Hbuf + (size_t)((((st + 1) & 1) * 2) + dir) * (Bdim * HDdim);

        if (st + 1 < Sdim) {   // prefetch next timestep's x rows for our tiles
            const int tn = dir ? (Sdim - 2 - st) : (st + 1);
            __builtin_prefetch(x_bf + ((size_t)(brow0 + r) * Sdim + tn) * Edim, 0, 0);
            __builtin_prefetch(x_bf + ((size_t)(brow1 + r) * Sdim + tn) * Edim, 0, 0);
        }

        v8f acc[2][4];
        v16bf a0, a1, na0, na1;

        // kk = 0: preload kk=1 first, then WMMA with C = inline 0
        a0 = load_frag(Xbase, brow0, SE, 0, r, hi);
        a1 = load_frag(Xbase, brow1, SE, 0, r, hi);
        na0 = load_frag(Xbase, brow0, SE, 32, r, hi);
        na1 = load_frag(Xbase, brow1, SE, 32, r, hi);
#pragma unroll
        for (int g = 0; g < 4; ++g) {
            v16bf bb = lds_frag(WL, (g * 4 + wq) * 16 + 0, lane);
            acc[0][g] = WMMA_BF16(a0, bb, vzero);
            acc[1][g] = WMMA_BF16(a1, bb, vzero);
        }
        a0 = na0; a1 = na1;

        // X part: kk = 1..7 (fully unrolled; all source selects compile-time)
#pragma unroll
        for (int kk = 1; kk < 8; ++kk) {
            if (kk < 7) {
                na0 = load_frag(Xbase, brow0, SE, (kk + 1) * 32, r, hi);
                na1 = load_frag(Xbase, brow1, SE, (kk + 1) * 32, r, hi);
            } else {
                na0 = load_frag(Hprev, brow0, HDdim, 0, r, hi);
                na1 = load_frag(Hprev, brow1, HDdim, 0, r, hi);
            }
#pragma unroll
            for (int g = 0; g < 4; ++g) {
                v16bf bb = lds_frag(WL, (g * 4 + wq) * 16 + kk, lane);
                acc[0][g] = WMMA_BF16(a0, bb, acc[0][g]);
                acc[1][g] = WMMA_BF16(a1, bb, acc[1][g]);
            }
            a0 = na0; a1 = na1;
        }

        // H part: kk = 8..15
#pragma unroll
        for (int kk = 8; kk < 16; ++kk) {
            if (kk < 15) {
                na0 = load_frag(Hprev, brow0, HDdim, (kk - 7) * 32, r, hi);
                na1 = load_frag(Hprev, brow1, HDdim, (kk - 7) * 32, r, hi);
            }
#pragma unroll
            for (int g = 0; g < 4; ++g) {
                v16bf bb = lds_frag(WL, (g * 4 + wq) * 16 + kk, lane);
                acc[0][g] = WMMA_BF16(a0, bb, acc[0][g]);
                acc[1][g] = WMMA_BF16(a1, bb, acc[1][g]);
            }
            a0 = na0; a1 = na1;
        }

        // gate nonlinearities + cell update, all in registers (bias here)
#pragma unroll
        for (int mt = 0; mt < 2; ++mt) {
#pragma unroll
            for (int v = 0; v < 8; ++v) {
                float iv = sigmoidf_(acc[mt][0][v] + bg[0]);
                float fv = sigmoidf_(acc[mt][1][v] + bg[1]);
                float gv = tanhf(acc[mt][2][v] + bg[2]);
                float ov = sigmoidf_(acc[mt][3][v] + bg[3]);
                float cv = fv * c[mt][v] + iv * gv;
                c[mt][v] = cv;
                float hv = ov * tanhf(cv);
                const int b = 32 * wm + 16 * mt + v + 8 * hi;  // D: m = v + 8*(lane/16)
                __bf16 hb = (__bf16)hv;
                Hnext[(size_t)b * HDdim + jcol] = hb;
                hs[((size_t)b * Sdim + t) * HDdim + jcol] = hb;
            }
        }

        // per-direction step barrier (monotonic counter; reset by prep_state)
        __threadfence();
        __syncthreads();
        if (tid == 0) {
            __hip_atomic_fetch_add(barD, 1u, __ATOMIC_RELEASE, __HIP_MEMORY_SCOPE_AGENT);
            const unsigned target = (unsigned)NBDIR * (unsigned)(st + 1);
            while (__hip_atomic_load(barD, __ATOMIC_ACQUIRE, __HIP_MEMORY_SCOPE_AGENT) < target) {
                __builtin_amdgcn_s_sleep(2);
            }
        }
        __syncthreads();
        __threadfence();
    }
}

// ---------------------------------------------------------------------------
// emissions[b,t,tag] = h_f . lin_w[tag,0:256] + h_b . lin_w[tag,256:512] + lin_b
// ---------------------------------------------------------------------------
__global__ __launch_bounds__(256)
void emissions_kernel(const __bf16* __restrict__ hs_f, const __bf16* __restrict__ hs_b,
                      const float* __restrict__ lin_w, const float* __restrict__ lin_b,
                      float* __restrict__ emis) {
    __shared__ float Wl[Tdim * 512];
    __shared__ float bl[Tdim];
    for (int i = threadIdx.x; i < Tdim * 512; i += blockDim.x) Wl[i] = lin_w[i];
    if (threadIdx.x < Tdim) bl[threadIdx.x] = lin_b[threadIdx.x];
    __syncthreads();

    const size_t bt = (size_t)blockIdx.x * blockDim.x + threadIdx.x;
    if (bt >= (size_t)Bdim * Sdim) return;
    const __bf16* pf = hs_f + bt * HDdim;
    const __bf16* pb = hs_b + bt * HDdim;

    float acc[Tdim];
#pragma unroll
    for (int tg = 0; tg < Tdim; ++tg) acc[tg] = bl[tg];

    for (int k = 0; k < HDdim; k += 8) {
        v8bf xf = *reinterpret_cast<const v8bf*>(pf + k);
        v8bf xb = *reinterpret_cast<const v8bf*>(pb + k);
        float xs[8], xs2[8];
#pragma unroll
        for (int u = 0; u < 8; ++u) { xs[u] = (float)xf[u]; xs2[u] = (float)xb[u]; }
#pragma unroll
        for (int tg = 0; tg < Tdim; ++tg) {
            float s = acc[tg];
#pragma unroll
            for (int u = 0; u < 8; ++u) s += xs[u]  * Wl[tg * 512 + k + u];
#pragma unroll
            for (int u = 0; u < 8; ++u) s += xs2[u] * Wl[tg * 512 + Edim + k + u];
            acc[tg] = s;
        }
    }
    float* out = emis + bt * Tdim;
#pragma unroll
    for (int tg = 0; tg < Tdim; ++tg) out[tg] = acc[tg];
}

// ---------------------------------------------------------------------------
// CRF log-likelihood: one thread per batch element; block-reduce -> scalar.
// ---------------------------------------------------------------------------
__global__ __launch_bounds__(128)
void crf_kernel(const float* __restrict__ emis, const int* __restrict__ y,
                const int* __restrict__ mask, const float* __restrict__ start_t,
                const float* __restrict__ end_t, const float* __restrict__ trans,
                float* __restrict__ out) {
    __shared__ float Tr[Tdim * Tdim], St[Tdim], En[Tdim];
    __shared__ float red[128];
    const int tid = threadIdx.x;
    if (tid < Tdim * Tdim) Tr[tid] = trans[tid];
    if (tid < Tdim) { St[tid] = start_t[tid]; En[tid] = end_t[tid]; }
    __syncthreads();

    const int b = tid;
    const float* em = emis + (size_t)b * Sdim * Tdim;
    const int* yb = y + (size_t)b * Sdim;
    const int* mb = mask + (size_t)b * Sdim;

    int yprev = yb[0];
    float score = St[yprev] + em[yprev];
    float alpha[Tdim];
#pragma unroll
    for (int j = 0; j < Tdim; ++j) alpha[j] = St[j] + em[j];

    for (int t = 1; t < Sdim; ++t) {
        if (mb[t] != 0) {
            const float* et = em + (size_t)t * Tdim;
            const int yt = yb[t];
            score += Tr[yprev * Tdim + yt] + et[yt];
            yprev = yt;
            float na[Tdim];
#pragma unroll
            for (int j = 0; j < Tdim; ++j) {
                float mx = -INFINITY;
#pragma unroll
                for (int i = 0; i < Tdim; ++i) mx = fmaxf(mx, alpha[i] + Tr[i * Tdim + j]);
                float s = 0.0f;
#pragma unroll
                for (int i = 0; i < Tdim; ++i) s += __expf(alpha[i] + Tr[i * Tdim + j] - mx);
                na[j] = mx + __logf(s) + et[j];
            }
#pragma unroll
            for (int j = 0; j < Tdim; ++j) alpha[j] = na[j];
        }
    }
    const float num = score + En[yprev];
    float mx = -INFINITY;
#pragma unroll
    for (int j = 0; j < Tdim; ++j) mx = fmaxf(mx, alpha[j] + En[j]);
    float s = 0.0f;
#pragma unroll
    for (int j = 0; j < Tdim; ++j) s += __expf(alpha[j] + En[j] - mx);
    const float den = mx + __logf(s);

    red[tid] = num - den;
    __syncthreads();
    for (int off = 64; off > 0; off >>= 1) {
        if (tid < off) red[tid] += red[tid + off];
        __syncthreads();
    }
    if (tid == 0) out[0] = red[0];
}

// ---------------------------------------------------------------------------
extern "C" void kernel_launch(void* const* d_in, const int* in_sizes, int n_in,
                              void* d_out, int out_size, void* d_ws, size_t ws_size,
                              hipStream_t stream) {
    const int*   X       = (const int*)d_in[0];
    const int*   y       = (const int*)d_in[1];
    const int*   mask    = (const int*)d_in[2];
    const float* emb     = (const float*)d_in[3];
    const float* w_ih_f  = (const float*)d_in[4];
    const float* w_hh_f  = (const float*)d_in[5];
    const float* b_ih_f  = (const float*)d_in[6];
    const float* b_hh_f  = (const float*)d_in[7];
    const float* w_ih_b  = (const float*)d_in[8];
    const float* w_hh_b  = (const float*)d_in[9];
    const float* b_ih_b  = (const float*)d_in[10];
    const float* b_hh_b  = (const float*)d_in[11];
    const float* lin_w   = (const float*)d_in[12];
    const float* lin_b   = (const float*)d_in[13];
    const float* start_t = (const float*)d_in[14];
    const float* end_t   = (const float*)d_in[15];
    const float* trans   = (const float*)d_in[16];
    float* out = (float*)d_out;

    // ---- workspace carve (256B aligned) ----
    char* ws = (char*)d_ws;
    size_t off = 0;
    auto carve = [&](size_t bytes) {
        void* p = ws + off;
        off += (bytes + 255) & ~(size_t)255;
        return p;
    };
    __bf16*   Wf     = (__bf16*)carve((size_t)NGATE * KCAT * 2);           // 1 MB
    __bf16*   Wb     = (__bf16*)carve((size_t)NGATE * KCAT * 2);           // 1 MB
    float*    bias_f = (float*)carve((size_t)NGATE * 4);
    float*    bias_b = (float*)carve((size_t)NGATE * 4);
    __bf16*   Hbuf   = (__bf16*)carve((size_t)2 * 2 * Bdim * HDdim * 2);   // 256 KB
    unsigned* bar    = (unsigned*)carve(256);                              // 2 counters, 64B apart
    __bf16*   x_bf   = (__bf16*)carve((size_t)Bdim * Sdim * Edim * 2);     // 32 MB
    __bf16*   hs_f   = (__bf16*)carve((size_t)Bdim * Sdim * HDdim * 2);    // 32 MB
    __bf16*   hs_b   = (__bf16*)carve((size_t)Bdim * Sdim * HDdim * 2);    // 32 MB
    float*    emis   = (float*)carve((size_t)Bdim * Sdim * Tdim * 4);      // 2.25 MB

    // ---- pipeline ----
    prep_weights<<<512, 256, 0, stream>>>(w_ih_f, w_hh_f, b_ih_f, b_hh_f,
                                          w_ih_b, w_hh_b, b_ih_b, b_hh_b,
                                          Wf, Wb, bias_f, bias_b);
    {
        const size_t total = (size_t)Bdim * Sdim * Edim / 4;
        prep_embed<<<(unsigned)((total + 255) / 256), 256, 0, stream>>>(X, emb, x_bf);
    }
    prep_state<<<(2 * 2 * Bdim * HDdim + 255) / 256, 256, 0, stream>>>(Hbuf, bar);

    const size_t ldsW = (size_t)4 * 4 * 16 * 512 * sizeof(__bf16);  // 256 KB
    lstm_recurrent<<<NBLK, 512, ldsW, stream>>>(x_bf, Wf, Wb, bias_f, bias_b,
                                                Hbuf, hs_f, hs_b, bar);

    emissions_kernel<<<(Bdim * Sdim + 255) / 256, 256, 0, stream>>>(hs_f, hs_b,
                                                                    lin_w, lin_b, emis);

    crf_kernel<<<1, 128, 0, stream>>>(emis, y, mask, start_t, end_t, trans, out);
}
